// CatalystGNNLayer_71519795413188
// MI455X (gfx1250) — matrix operand order repaired and checked by reading, not compiled, not run on hardware
//
#include <hip/hip_runtime.h>

#define N_NODES 50000
#define N_EDGES 800000
#define HID 128

typedef __attribute__((ext_vector_type(16))) __bf16 v16bf;
typedef __attribute__((ext_vector_type(8)))  float  v8f;

// fp32 -> bf16 bits, round-to-nearest-even
__device__ __forceinline__ unsigned short f2bf(float f) {
  unsigned u = __builtin_bit_cast(unsigned, f);
  u += 0x7FFFu + ((u >> 16) & 1u);
  return (unsigned short)(u >> 16);
}

union FragU { uint4 q[2]; v16bf v; };

__device__ __forceinline__ v8f vzero() {
  v8f z;
#pragma unroll
  for (int i = 0; i < 8; ++i) z[i] = 0.0f;
  return z;
}

// B fragment (32x16 bf16, KxN): lane 0-15 = col lane, K 0..15; lane 16-31 = col lane-16, K 16..31
__device__ __forceinline__ v16bf load_b_frag(const unsigned short* wf, int frag, int lane) {
  const uint4* p = (const uint4*)(wf + (((size_t)frag * 32 + lane) << 4));
  FragU t; t.q[0] = p[0]; t.q[1] = p[1];
  return t.v;
}

// A fragment (16x32 bf16, MxK) from LDS row-major [rows][stride]:
// lane<16: M=lane, K k0+0..15 ; lane>=16: M=lane-16, K k0+16..31
__device__ __forceinline__ v16bf load_a_frag(const unsigned short* lds, int stride,
                                             int mbase, int k0, int lane) {
  int m  = mbase + (lane & 15);
  int kq = k0 + ((lane >> 4) << 4);
  const unsigned short* r = lds + m * stride + kq;
  FragU t;
  t.q[0] = *(const uint4*)(r);
  t.q[1] = *(const uint4*)(r + 8);
  return t.v;
}

#define WMMA_BF16(A, B, C) \
  __builtin_amdgcn_wmma_f32_16x16x32_bf16(false, (A), false, (B), (short)0, (C), false, false)

// Repack fp32 weights (KxN row-major) into bf16 WMMA B-fragments.
// Layout in ws: [w1: 64 frags][w2: 32][u1: 64][u2: 32], frag = kt*8+nt, 512 bf16 each.
__global__ __launch_bounds__(256) void convert_weights(
    const float* __restrict__ w1, const float* __restrict__ w2,
    const float* __restrict__ u1, const float* __restrict__ u2,
    unsigned short* __restrict__ wf) {
  int idx = blockIdx.x * 256 + threadIdx.x;           // 0 .. 98303
  const float* src; int local;
  if (idx < 32768)       { src = w1; local = idx; }
  else if (idx < 49152)  { src = w2; local = idx - 32768; }
  else if (idx < 81920)  { src = u1; local = idx - 49152; }
  else                   { src = u2; local = idx - 81920; }
  int frag   = local >> 9;
  int within = local & 511;
  int lane   = within >> 4;
  int i      = within & 15;
  int kt = frag >> 3, nt = frag & 7;
  int k = kt * 32 + ((lane >> 4) << 4) + i;           // lanes>=16 hold K 16..31
  int n = nt * 16 + (lane & 15);
  wf[idx] = f2bf(src[k * HID + n]);
}

constexpr int ASTR = 264, HSTR = 136;                 // padded LDS strides (bank-spread)
constexpr int ROWS = 64;                              // rows (edges / nodes) per block

// ---- Edge message kernel: 64 edges / block, 8 waves, each wave owns a 32x32 tile ----
__global__ __launch_bounds__(256) void edge_kernel(
    const float* __restrict__ x, const int* __restrict__ ei,
    const unsigned short* __restrict__ wf,
    const float* __restrict__ b1, const float* __restrict__ b2,
    float* __restrict__ msgs) {
  __shared__ unsigned short sA[ROWS * ASTR];
  __shared__ unsigned short sH[ROWS * HSTR];
  __shared__ int sSrc[ROWS];
  __shared__ int sDst[ROWS];

  const int t  = threadIdx.x;
  const int e0 = blockIdx.x * ROWS;
  if (t < ROWS) {
    sSrc[t] = ei[(size_t)(e0 + t) * 2];
    sDst[t] = ei[(size_t)(e0 + t) * 2 + 1];
  }
  __syncthreads();
  {
    const int c = t & 127;
    const int half = t >> 7;                          // 0: x[src] cols, 1: x[dst] cols
#pragma unroll 4
    for (int r = 0; r < ROWS; ++r) {
      int node = half ? sDst[r] : sSrc[r];
      sA[r * ASTR + half * 128 + c] = f2bf(x[(size_t)node * HID + c]);
    }
  }
  __syncthreads();

  const int l      = t & 31;
  const int w      = t >> 5;
  const int mbase0 = (w >> 2) << 5;                   // 0 or 32
  const int mbase1 = mbase0 + 16;
  const int nt0    = (w & 3) << 1;                    // 0,2,4,6

  // Layer 1: 256 -> 128, ReLU
  v8f c00 = vzero(), c01 = vzero(), c10 = vzero(), c11 = vzero();
#pragma unroll
  for (int kt = 0; kt < 8; ++kt) {
    v16bf a0 = load_a_frag(sA, ASTR, mbase0, kt * 32, l);
    v16bf a1 = load_a_frag(sA, ASTR, mbase1, kt * 32, l);
    v16bf bA = load_b_frag(wf, kt * 8 + nt0, l);
    v16bf bB = load_b_frag(wf, kt * 8 + nt0 + 1, l);
    c00 = WMMA_BF16(a0, bA, c00);
    c01 = WMMA_BF16(a0, bB, c01);
    c10 = WMMA_BF16(a1, bA, c10);
    c11 = WMMA_BF16(a1, bB, c11);
  }
  {
    const int colA = (nt0 << 4) + (l & 15);
    const int colB = colA + 16;
    const float biA = b1[colA], biB = b1[colB];
    const int r0 = mbase0 + ((l >> 4) << 3);
    const int r1 = mbase1 + ((l >> 4) << 3);
#pragma unroll
    for (int r = 0; r < 8; ++r) {
      float v00 = c00[r] + biA; v00 = v00 > 0.0f ? v00 : 0.0f;
      float v01 = c01[r] + biB; v01 = v01 > 0.0f ? v01 : 0.0f;
      float v10 = c10[r] + biA; v10 = v10 > 0.0f ? v10 : 0.0f;
      float v11 = c11[r] + biB; v11 = v11 > 0.0f ? v11 : 0.0f;
      sH[(r0 + r) * HSTR + colA] = f2bf(v00);
      sH[(r0 + r) * HSTR + colB] = f2bf(v01);
      sH[(r1 + r) * HSTR + colA] = f2bf(v10);
      sH[(r1 + r) * HSTR + colB] = f2bf(v11);
    }
  }
  __syncthreads();

  // Layer 2: 128 -> 128, then scatter-add
  const unsigned short* wf2 = wf + 64 * 512;
  c00 = vzero(); c01 = vzero(); c10 = vzero(); c11 = vzero();
#pragma unroll
  for (int kt = 0; kt < 4; ++kt) {
    v16bf a0 = load_a_frag(sH, HSTR, mbase0, kt * 32, l);
    v16bf a1 = load_a_frag(sH, HSTR, mbase1, kt * 32, l);
    v16bf bA = load_b_frag(wf2, kt * 8 + nt0, l);
    v16bf bB = load_b_frag(wf2, kt * 8 + nt0 + 1, l);
    c00 = WMMA_BF16(a0, bA, c00);
    c01 = WMMA_BF16(a0, bB, c01);
    c10 = WMMA_BF16(a1, bA, c10);
    c11 = WMMA_BF16(a1, bB, c11);
  }
  {
    const int colA = (nt0 << 4) + (l & 15);
    const int colB = colA + 16;
    const float biA = b2[colA], biB = b2[colB];
    const int r0 = mbase0 + ((l >> 4) << 3);
    const int r1 = mbase1 + ((l >> 4) << 3);
#pragma unroll
    for (int r = 0; r < 8; ++r) {
      int d0 = sDst[r0 + r];
      int d1 = sDst[r1 + r];
      unsafeAtomicAdd(&msgs[(size_t)d0 * HID + colA], c00[r] + biA);
      unsafeAtomicAdd(&msgs[(size_t)d0 * HID + colB], c01[r] + biB);
      unsafeAtomicAdd(&msgs[(size_t)d1 * HID + colA], c10[r] + biA);
      unsafeAtomicAdd(&msgs[(size_t)d1 * HID + colB], c11[r] + biB);
    }
  }
}

// ---- Node update kernel: 64 nodes / block, same wave tiling ----
__global__ __launch_bounds__(256) void node_kernel(
    const float* __restrict__ x, const float* __restrict__ msgs,
    const unsigned short* __restrict__ wf,               // full frag base
    const float* __restrict__ b1, const float* __restrict__ b2,
    float* __restrict__ out) {
  __shared__ unsigned short sA[ROWS * ASTR];
  __shared__ unsigned short sH[ROWS * HSTR];

  const int t  = threadIdx.x;
  const int i0 = blockIdx.x * ROWS;
  {
    const int c = t & 127;
    const int half = t >> 7;                          // 0: x, 1: messages
#pragma unroll 4
    for (int r = 0; r < ROWS; ++r) {
      int i = i0 + r; if (i >= N_NODES) i = N_NODES - 1;
      const float* srcp = half ? (msgs + (size_t)i * HID) : (x + (size_t)i * HID);
      sA[r * ASTR + half * 128 + c] = f2bf(srcp[c]);
    }
  }
  __syncthreads();

  const int l      = t & 31;
  const int w      = t >> 5;
  const int mbase0 = (w >> 2) << 5;
  const int mbase1 = mbase0 + 16;
  const int nt0    = (w & 3) << 1;

  const unsigned short* wfU1 = wf + 96 * 512;          // after w1(64)+w2(32) frags
  const unsigned short* wfU2 = wf + 160 * 512;

  v8f c00 = vzero(), c01 = vzero(), c10 = vzero(), c11 = vzero();
#pragma unroll
  for (int kt = 0; kt < 8; ++kt) {
    v16bf a0 = load_a_frag(sA, ASTR, mbase0, kt * 32, l);
    v16bf a1 = load_a_frag(sA, ASTR, mbase1, kt * 32, l);
    v16bf bA = load_b_frag(wfU1, kt * 8 + nt0, l);
    v16bf bB = load_b_frag(wfU1, kt * 8 + nt0 + 1, l);
    c00 = WMMA_BF16(a0, bA, c00);
    c01 = WMMA_BF16(a0, bB, c01);
    c10 = WMMA_BF16(a1, bA, c10);
    c11 = WMMA_BF16(a1, bB, c11);
  }
  {
    const int colA = (nt0 << 4) + (l & 15);
    const int colB = colA + 16;
    const float biA = b1[colA], biB = b1[colB];
    const int r0 = mbase0 + ((l >> 4) << 3);
    const int r1 = mbase1 + ((l >> 4) << 3);
#pragma unroll
    for (int r = 0; r < 8; ++r) {
      float v00 = c00[r] + biA; v00 = v00 > 0.0f ? v00 : 0.0f;
      float v01 = c01[r] + biB; v01 = v01 > 0.0f ? v01 : 0.0f;
      float v10 = c10[r] + biA; v10 = v10 > 0.0f ? v10 : 0.0f;
      float v11 = c11[r] + biB; v11 = v11 > 0.0f ? v11 : 0.0f;
      sH[(r0 + r) * HSTR + colA] = f2bf(v00);
      sH[(r0 + r) * HSTR + colB] = f2bf(v01);
      sH[(r1 + r) * HSTR + colA] = f2bf(v10);
      sH[(r1 + r) * HSTR + colB] = f2bf(v11);
    }
  }
  __syncthreads();

  c00 = vzero(); c01 = vzero(); c10 = vzero(); c11 = vzero();
#pragma unroll
  for (int kt = 0; kt < 4; ++kt) {
    v16bf a0 = load_a_frag(sH, HSTR, mbase0, kt * 32, l);
    v16bf a1 = load_a_frag(sH, HSTR, mbase1, kt * 32, l);
    v16bf bA = load_b_frag(wfU2, kt * 8 + nt0, l);
    v16bf bB = load_b_frag(wfU2, kt * 8 + nt0 + 1, l);
    c00 = WMMA_BF16(a0, bA, c00);
    c01 = WMMA_BF16(a0, bB, c01);
    c10 = WMMA_BF16(a1, bA, c10);
    c11 = WMMA_BF16(a1, bB, c11);
  }
  {
    const int colA = (nt0 << 4) + (l & 15);
    const int colB = colA + 16;
    const float biA = b2[colA], biB = b2[colB];
    const int r0 = mbase0 + ((l >> 4) << 3);
    const int r1 = mbase1 + ((l >> 4) << 3);
#pragma unroll
    for (int r = 0; r < 8; ++r) {
      int row0 = i0 + r0 + r;
      int row1 = i0 + r1 + r;
      if (row0 < N_NODES) {
        out[(size_t)row0 * HID + colA] = c00[r] + biA;
        out[(size_t)row0 * HID + colB] = c01[r] + biB;
      }
      if (row1 < N_NODES) {
        out[(size_t)row1 * HID + colA] = c10[r] + biA;
        out[(size_t)row1 * HID + colB] = c11[r] + biB;
      }
    }
  }
}

extern "C" void kernel_launch(void* const* d_in, const int* in_sizes, int n_in,
                              void* d_out, int out_size, void* d_ws, size_t ws_size,
                              hipStream_t stream) {
  const float* x   = (const float*)d_in[0];
  const int*   ei  = (const int*)d_in[1];
  const float* mw1 = (const float*)d_in[2];
  const float* mb1 = (const float*)d_in[3];
  const float* mw2 = (const float*)d_in[4];
  const float* mb2 = (const float*)d_in[5];
  const float* uw1 = (const float*)d_in[6];
  const float* ub1 = (const float*)d_in[7];
  const float* uw2 = (const float*)d_in[8];
  const float* ub2 = (const float*)d_in[9];

  unsigned short* wf = (unsigned short*)d_ws;                       // 196608 B of bf16 frags
  float* msgs = (float*)((char*)d_ws + 196608);                     // 50000*128 f32 accumulator

  hipMemsetAsync(msgs, 0, (size_t)N_NODES * HID * sizeof(float), stream);
  convert_weights<<<384, 256, 0, stream>>>(mw1, mw2, uw1, uw2, wf);
  edge_kernel<<<N_EDGES / ROWS, 256, 0, stream>>>(x, ei, wf, mb1, mb2, msgs);
  node_kernel<<<(N_NODES + ROWS - 1) / ROWS, 256, 0, stream>>>(x, msgs, wf, ub1, ub2,
                                                               (float*)d_out);
}
